// Block_49340584297221
// MI455X (gfx1250) — matrix operand order repaired and checked by reading, not compiled, not run on hardware
//
#include <hip/hip_runtime.h>
#include <math.h>

#define NNODES 20000
#define NEDGES 262144

typedef __attribute__((ext_vector_type(16))) __bf16 bf16x16;
typedef __attribute__((ext_vector_type(8)))  float  f32x8;

#define DEVI static __device__ __forceinline__

// ---- packed-weight fragment indices in d_ws (each frag = 32 lanes x 16 bf16)
#define F_W1A(i, kc)  ((i) * 4 + (kc))        // 12 frags
#define F_W1V(t, kc)  (12 + (t) * 2 + (kc))   // 8
#define F_W1D(t)      (20 + (t))              // 4
#define F_W2(i)       (24 + (i))              // 11
#define F_WOA(nt, kc) (35 + (nt) * 2 + (kc))  // 16
#define F_WOV(nt, kc) (51 + (nt) * 2 + (kc))  // 8
#define F_WOD(nt, kc) (59 + (nt) * 2 + (kc))  // 4
#define NFRAGS 63

DEVI f32x8 vzero8() {
  f32x8 z;
#pragma unroll
  for (int k = 0; k < 8; ++k) z[k] = 0.0f;
  return z;
}

DEVI f32x8 wmma_bf16(bf16x16 a, bf16x16 b, f32x8 c) {
  // 8-arg form: (neg_a, A, neg_b, B, c_mod, C, reuse_a, reuse_b)
  return __builtin_amdgcn_wmma_f32_16x16x32_bf16(false, a, false, b, (short)0, c,
                                                 false, false);
}

DEVI float bperm_f(float v, int srclane) {
  return __int_as_float(
      __builtin_amdgcn_ds_bpermute(srclane << 2, __float_as_int(v)));
}

// ---- fragment builders -----------------------------------------------------
// A (16x32 bf16): lane<16 owns row lane, K runs {o..o+15}, o=hi*16
DEVI bf16x16 afrag_g(const float* rowbase, int lane) {
  const int o = ((lane >> 4) & 1) * 16;
  bf16x16 r;
#pragma unroll
  for (int k = 0; k < 8; ++k) r[k] = (__bf16)rowbase[o + k];
#pragma unroll
  for (int k = 0; k < 8; ++k) r[8 + k] = (__bf16)rowbase[o + 8 + k];
  return r;
}

DEVI bf16x16 afrag_gs(const float* rowbase, int stride, int lane) {
  const int o = ((lane >> 4) & 1) * 16;
  bf16x16 r;
#pragma unroll
  for (int k = 0; k < 8; ++k) r[k] = (__bf16)rowbase[(o + k) * stride];
#pragma unroll
  for (int k = 0; k < 8; ++k) r[8 + k] = (__bf16)rowbase[(o + 8 + k) * stride];
  return r;
}

// Pre-packed B fragment: one aligned 32B vector load per lane
DEVI bf16x16 bfrag_ws(const __bf16* wsb, int fragIdx, int lane) {
  return *(const bf16x16*)(wsb + ((size_t)fragIdx * 32 + lane) * 16);
}

// rad (16x8) as A-frag zero-padded to K=32 (each of lanes 0..15 owns its row)
DEVI bf16x16 radfrag_make(const float* rad8, int lane) {
  const bool lo = lane < 16;
  bf16x16 r;
#pragma unroll
  for (int k = 0; k < 8; ++k) r[k] = (__bf16)(lo ? rad8[k] : 0.0f);
#pragma unroll
  for (int k = 0; k < 8; ++k) r[8 + k] = (__bf16)0.0f;
  return r;
}

// Re-layout: two staged LDS tiles [edge][rank] -> bf16 A-frag, per-half scales
DEVI bf16x16 afrag_lds(const float* tX, float sX, const float* tY, float sY,
                       int lane) {
  const int mrow = lane & 15;
  const int hi = (lane >> 4) & 1;
  const float* tt = hi ? tY : tX;
  const float ss = hi ? sY : sX;
  bf16x16 r;
#pragma unroll
  for (int k = 0; k < 16; ++k) r[k] = (__bf16)(tt[mrow * 16 + k] * ss);
  return r;
}

// Store C-layout accumulator tile to LDS as [edge][rank]
DEVI void tstore(float* tile, f32x8 v, int lane) {
  const int n = lane & 15;
  const int mb = ((lane >> 4) & 1) * 8;
#pragma unroll
  for (int g = 0; g < 8; ++g) tile[(mb + g) * 16 + n] = v[g];
}

DEVI void atomic_add_f32(float* p, float v) {
  __hip_atomic_fetch_add(p, v, __ATOMIC_RELAXED, __HIP_MEMORY_SCOPE_AGENT);
}

// ---- kernels ---------------------------------------------------------------
__global__ void zero_kernel(float* __restrict__ p, int n) {
  int i = blockIdx.x * blockDim.x + threadIdx.x;
  if (i < n) p[i] = 0.0f;
}

// One block per fragment: convert + swizzle weights into WMMA B layout (bf16)
__global__ __launch_bounds__(32) void pack_weights_kernel(
    const float* __restrict__ W1_a, const float* __restrict__ W1_v,
    const float* __restrict__ W1_d, const float* __restrict__ W2,
    const float* __restrict__ Wo_a, const float* __restrict__ Wo_v,
    const float* __restrict__ Wo_d, __bf16* __restrict__ wsb) {
  const int b = blockIdx.x;
  const int lane = threadIdx.x;
  const int col = lane & 15;
  const int hi = (lane >> 4) & 1;
  float v[16];
#pragma unroll
  for (int k = 0; k < 16; ++k) v[k] = 0.0f;

  if (b < 12) {                      // W1_a (3,16,128): i, kc(4)
    const int i = b >> 2, kc = b & 3;
    const float* p = W1_a + ((size_t)(i * 16 + col)) * 128 + kc * 32 + hi * 16;
#pragma unroll
    for (int k = 0; k < 16; ++k) v[k] = p[k];
  } else if (b < 20) {               // W1_v (4,16,64): t, kc(2)
    const int t = (b - 12) >> 1, kc = (b - 12) & 1;
    const float* p = W1_v + ((size_t)(t * 16 + col)) * 64 + kc * 32 + hi * 16;
#pragma unroll
    for (int k = 0; k < 16; ++k) v[k] = p[k];
  } else if (b < 24) {               // W1_d (4,16,32): t
    const int t = b - 20;
    const float* p = W1_d + ((size_t)(t * 16 + col)) * 32 + hi * 16;
#pragma unroll
    for (int k = 0; k < 16; ++k) v[k] = p[k];
  } else if (b < 35) {               // W2 (11,16,8): K valid 0..7, lanes<16 only
    const int i = b - 24;
    if (hi == 0) {
#pragma unroll
      for (int k = 0; k < 8; ++k) v[k] = W2[(size_t)i * 128 + col * 8 + k];
    }
  } else if (b < 51) {               // Wo_a (3,128,16): nt(8), kc(2); t=3 -> 0
    const int f = b - 35, nt = f >> 1, kc = f & 1, t = kc * 2 + hi;
    if (t < 3) {
      const float* p = Wo_a + ((size_t)(t * 128 + nt * 16 + col)) * 16;
#pragma unroll
      for (int k = 0; k < 16; ++k) v[k] = p[k];
    }
  } else if (b < 59) {               // Wo_v (4,64,16): nt(4), kc(2)
    const int f = b - 51, nt = f >> 1, kc = f & 1, t = kc * 2 + hi;
    const float* p = Wo_v + ((size_t)(t * 64 + nt * 16 + col)) * 16;
#pragma unroll
    for (int k = 0; k < 16; ++k) v[k] = p[k];
  } else {                           // Wo_d (4,32,16): nt(2), kc(2)
    const int f = b - 59, nt = f >> 1, kc = f & 1, t = kc * 2 + hi;
    const float* p = Wo_d + ((size_t)(t * 32 + nt * 16 + col)) * 16;
#pragma unroll
    for (int k = 0; k < 16; ++k) v[k] = p[k];
  }

  __bf16* outp = wsb + ((size_t)b * 32 + lane) * 16;
#pragma unroll
  for (int k = 0; k < 16; ++k) outp[k] = (__bf16)v[k];
}

__global__ __launch_bounds__(32) void equi_mp_kernel(
    const float* __restrict__ r_ij, const float* __restrict__ x_a,
    const float* __restrict__ x_v, const float* __restrict__ x_d,
    const __bf16* __restrict__ wsb, const int* __restrict__ src,
    const int* __restrict__ dst, float* __restrict__ out) {
  __shared__ float sT[28 * 256];  // 27 staged tiles + 1 zero tile (28 KB)
  const int lane = threadIdx.x;
  const int e = blockIdx.x * 16 + (lane & 15);

  tstore(sT + 27 * 256, vzero8(), lane);  // zero-pad tile

  // ---- per-edge scalar features (all lanes mirror edge lane&15)
  const float rx = r_ij[(size_t)e * 3 + 0];
  const float ry = r_ij[(size_t)e * 3 + 1];
  const float rz = r_ij[(size_t)e * 3 + 2];
  const float x_sq = (rx * rx + ry * ry + rz * rz) * 0.2f;  // /R0
  const float env = fmaxf(1.0f - x_sq, 0.0f);
  const float sq = sqrtf(x_sq);
  float rad[8];
#pragma unroll
  for (int k = 0; k < 8; ++k)
    rad[k] = __cosf(3.14159265358979f * (float)k * sq) * env;
  const float vx = rx * 3.4f, vy = ry * 3.4f, vz = rz * 3.4f;
  const float nn = sqrtf(vx * vx + vy * vy + vz * vz);
  const float fac = (2.0f / (1.0f + __expf(-nn)) - 1.0f) / (nn + 1e-6f);
  const float rv0 = vx * fac, rv1 = vy * fac, rv2 = vz * fac;
  const int dstRow = dst[e];
  const int srcRow = src[e];

  const bf16x16 radf = radfrag_make(rad, lane);

  // Broadcast rv / src into WMMA C-layout (edge = g + 8*(lane>>4))
  const int hb8 = ((lane >> 4) & 1) * 8;
  f32x8 rvb[3];
  int srow[8];
#pragma unroll
  for (int g = 0; g < 8; ++g) {
    const int sl = hb8 + g;
    rvb[0][g] = bperm_f(rv0, sl);
    rvb[1][g] = bperm_f(rv1, sl);
    rvb[2][g] = bperm_f(rv2, sl);
    srow[g] = __builtin_amdgcn_ds_bpermute(sl << 2, srcRow);
  }

  // ---- Phase A: scalar-channel projections (pa000/011/022), K=128
  {
    const float* xaRow = x_a + (size_t)dstRow * 128;
    f32x8 pa0 = vzero8(), pa1 = vzero8(), pa2 = vzero8();
#pragma unroll
    for (int kc = 0; kc < 4; ++kc) {
      const bf16x16 af = afrag_g(xaRow + kc * 32, lane);
      pa0 = wmma_bf16(af, bfrag_ws(wsb, F_W1A(0, kc), lane), pa0);
      pa1 = wmma_bf16(af, bfrag_ws(wsb, F_W1A(1, kc), lane), pa1);
      pa2 = wmma_bf16(af, bfrag_ws(wsb, F_W1A(2, kc), lane), pa2);
    }
    f32x8 s0 = wmma_bf16(radf, bfrag_ws(wsb, F_W2(0), lane), vzero8());
    tstore(sT + 0 * 256, pa0 * s0, lane);                 // Ta0 = pa000*s0
    f32x8 s3 = wmma_bf16(radf, bfrag_ws(wsb, F_W2(3), lane), vzero8());
    tstore(sT + 3 * 256, pa1 * s3, lane);                 // pa011*s3
    f32x8 s7 = wmma_bf16(radf, bfrag_ws(wsb, F_W2(7), lane), vzero8());
    tstore(sT + 11 * 256, pa2 * s7, lane);                // pa022*s7
  }

  // ---- Phase V: vector-channel projections, 3 spatial comps, K=64
  {
    const float* xvRow = x_v + (size_t)dstRow * 192;
    const f32x8 s4 = wmma_bf16(radf, bfrag_ws(wsb, F_W2(4), lane), vzero8());
    const f32x8 s8 = wmma_bf16(radf, bfrag_ws(wsb, F_W2(8), lane), vzero8());
    f32x8 acc110 = vzero8(), acc121 = vzero8();
#pragma unroll
    for (int i = 0; i < 3; ++i) {
      f32x8 p101 = vzero8(), p110 = vzero8(), p112 = vzero8(), p121 = vzero8();
#pragma unroll
      for (int kc = 0; kc < 2; ++kc) {
        const bf16x16 af = afrag_gs(xvRow + kc * 96 + i, 3, lane);
        p101 = wmma_bf16(af, bfrag_ws(wsb, F_W1V(0, kc), lane), p101);
        p110 = wmma_bf16(af, bfrag_ws(wsb, F_W1V(1, kc), lane), p110);
        p112 = wmma_bf16(af, bfrag_ws(wsb, F_W1V(2, kc), lane), p112);
        p121 = wmma_bf16(af, bfrag_ws(wsb, F_W1V(3, kc), lane), p121);
      }
      tstore(sT + (4 + i) * 256, p101 * s4, lane);        // pv101_i*s4
      tstore(sT + (12 + i) * 256, p112 * s8, lane);       // pv112_i*s8
      acc110 += p110 * rvb[i];                            // Σ_i pv110_i rv_i
      acc121 += p121 * rvb[i];                            // Σ_i pv121_i rv_i
    }
    const f32x8 s1 = wmma_bf16(radf, bfrag_ws(wsb, F_W2(1), lane), vzero8());
    tstore(sT + 1 * 256, acc110 * s1, lane);              // Ta1
    const f32x8 s5 = wmma_bf16(radf, bfrag_ws(wsb, F_W2(5), lane), vzero8());
    tstore(sT + 7 * 256, acc121 * s5, lane);              // V121c*s5
  }

  // ---- Phase D: tensor-channel projections, 9 comps, K=32
  {
    const float* xdRow = x_d + (size_t)dstRow * 288;
    const f32x8 s9 = wmma_bf16(radf, bfrag_ws(wsb, F_W2(9), lane), vzero8());
    bf16x16 bd[4];
#pragma unroll
    for (int t = 0; t < 4; ++t) bd[t] = bfrag_ws(wsb, F_W1D(t), lane);
    f32x8 q[3], mm[3], acc220 = vzero8();
#pragma unroll
    for (int i = 0; i < 3; ++i) { q[i] = vzero8(); mm[i] = vzero8(); }
#pragma unroll
    for (int i = 0; i < 3; ++i) {
#pragma unroll
      for (int j = 0; j < 3; ++j) {
        const int c = 3 * i + j;
        const bf16x16 af = afrag_gs(xdRow + c, 9, lane);
        f32x8 p202 = wmma_bf16(af, bd[0], vzero8());
        f32x8 p211 = wmma_bf16(af, bd[1], vzero8());
        f32x8 p220 = wmma_bf16(af, bd[2], vzero8());
        f32x8 p222 = wmma_bf16(af, bd[3], vzero8());
        tstore(sT + (15 + c) * 256, p202 * s9, lane);     // pd202_ij*s9
        q[i] += p211 * rvb[j];                            // Σ_j pd211_ij rv_j
        acc220 += p220 * (rvb[i] * rvb[j]);               // ΣΣ pd220 rv rv
        mm[i] += p222 * rvb[j];                           // Σ_k pd222_ik rv_k
      }
    }
    const f32x8 s6 = wmma_bf16(radf, bfrag_ws(wsb, F_W2(6), lane), vzero8());
    const f32x8 s2 = wmma_bf16(radf, bfrag_ws(wsb, F_W2(2), lane), vzero8());
    const f32x8 s10 = wmma_bf16(radf, bfrag_ws(wsb, F_W2(10), lane), vzero8());
#pragma unroll
    for (int i = 0; i < 3; ++i) {
      tstore(sT + (8 + i) * 256, q[i] * s6, lane);        // Q_i
      tstore(sT + (24 + i) * 256, mm[i] * s10, lane);     // M_i
    }
    tstore(sT + 2 * 256, acc220 * s2, lane);              // Ta2
  }

  __syncthreads();

  const float* zt = sT + 27 * 256;
  const float rvo[3] = {rv0, rv1, rv2};

  // ---- psi_a = [Ta0|Ta1|Ta2|0] @ WoA  -> atomic scatter into B_a
  {
    const bf16x16 a0 = afrag_lds(sT + 0 * 256, 1.0f, sT + 1 * 256, 1.0f, lane);
    const bf16x16 a1 = afrag_lds(sT + 2 * 256, 1.0f, zt, 0.0f, lane);
#pragma unroll
    for (int nt = 0; nt < 8; ++nt) {
      f32x8 acc = wmma_bf16(a0, bfrag_ws(wsb, F_WOA(nt, 0), lane), vzero8());
      acc = wmma_bf16(a1, bfrag_ws(wsb, F_WOA(nt, 1), lane), acc);
      const int col = nt * 16 + (lane & 15);
#pragma unroll
      for (int g = 0; g < 8; ++g)
        atomic_add_f32(out + (size_t)srow[g] * 128 + col, acc[g]);
    }
  }

  // ---- psi_v comp j = [pa011*s3*rv_j | pv101_j*s4 | V121cs*rv_j | Q_j] @ WoV
  {
    float* outV = out + (size_t)NNODES * 128;
    bf16x16 wov[8];  // hoisted: reused for all 3 spatial comps
#pragma unroll
    for (int f = 0; f < 8; ++f) bf16x16 dummy;  // (placeholder removed below)
#pragma unroll
    for (int nt = 0; nt < 4; ++nt) {
      wov[nt * 2 + 0] = bfrag_ws(wsb, F_WOV(nt, 0), lane);
      wov[nt * 2 + 1] = bfrag_ws(wsb, F_WOV(nt, 1), lane);
    }
#pragma unroll
    for (int j = 0; j < 3; ++j) {
      const float rj = rvo[j];
      const bf16x16 a0 = afrag_lds(sT + 3 * 256, rj, sT + (4 + j) * 256, 1.0f, lane);
      const bf16x16 a1 = afrag_lds(sT + 7 * 256, rj, sT + (8 + j) * 256, 1.0f, lane);
#pragma unroll
      for (int nt = 0; nt < 4; ++nt) {
        f32x8 acc = wmma_bf16(a0, wov[nt * 2 + 0], vzero8());
        acc = wmma_bf16(a1, wov[nt * 2 + 1], acc);
        const int col = nt * 16 + (lane & 15);
#pragma unroll
        for (int g = 0; g < 8; ++g)
          atomic_add_f32(outV + ((size_t)srow[g] * 64 + col) * 3 + j, acc[g]);
      }
    }
  }

  // ---- psi_d comp (i,j) = [pa022*s7*ri*rj | U_i*rj | D_ij | M_i*rj] @ WoD
  {
    float* outD = out + (size_t)NNODES * (128 + 192);
    bf16x16 wod[4];  // hoisted: reused for all 9 tensor comps
#pragma unroll
    for (int nt = 0; nt < 2; ++nt) {
      wod[nt * 2 + 0] = bfrag_ws(wsb, F_WOD(nt, 0), lane);
      wod[nt * 2 + 1] = bfrag_ws(wsb, F_WOD(nt, 1), lane);
    }
#pragma unroll
    for (int i = 0; i < 3; ++i) {
#pragma unroll
      for (int j = 0; j < 3; ++j) {
        const int c = 3 * i + j;
        const float ri = rvo[i], rj = rvo[j];
        const bf16x16 a0 = afrag_lds(sT + 11 * 256, ri * rj, sT + (12 + i) * 256, rj, lane);
        const bf16x16 a1 = afrag_lds(sT + (15 + c) * 256, 1.0f, sT + (24 + i) * 256, rj, lane);
#pragma unroll
        for (int nt = 0; nt < 2; ++nt) {
          f32x8 acc = wmma_bf16(a0, wod[nt * 2 + 0], vzero8());
          acc = wmma_bf16(a1, wod[nt * 2 + 1], acc);
          const int col = nt * 16 + (lane & 15);
#pragma unroll
          for (int g = 0; g < 8; ++g)
            atomic_add_f32(outD + ((size_t)srow[g] * 32 + col) * 9 + c, acc[g]);
        }
      }
    }
  }
}

extern "C" void kernel_launch(void* const* d_in, const int* in_sizes, int n_in,
                              void* d_out, int out_size, void* d_ws,
                              size_t ws_size, hipStream_t stream) {
  const float* r_ij = (const float*)d_in[0];
  const float* x_a  = (const float*)d_in[1];
  const float* x_v  = (const float*)d_in[2];
  const float* x_d  = (const float*)d_in[3];
  const float* W1_a = (const float*)d_in[4];
  const float* W1_v = (const float*)d_in[5];
  const float* W1_d = (const float*)d_in[6];
  const float* W2   = (const float*)d_in[7];
  const float* Wo_a = (const float*)d_in[8];
  const float* Wo_v = (const float*)d_in[9];
  const float* Wo_d = (const float*)d_in[10];
  const int* src = (const int*)d_in[11];
  const int* dst = (const int*)d_in[12];
  float* out = (float*)d_out;
  __bf16* wsb = (__bf16*)d_ws;

  pack_weights_kernel<<<NFRAGS, 32, 0, stream>>>(W1_a, W1_v, W1_d, W2, Wo_a,
                                                 Wo_v, Wo_d, wsb);
  zero_kernel<<<(out_size + 511) / 512, 512, 0, stream>>>(out, out_size);
  equi_mp_kernel<<<NEDGES / 16, 32, 0, stream>>>(r_ij, x_a, x_v, x_d, wsb, src,
                                                 dst, out);
}
